// NearestNeighborAttention_86638080295016
// MI455X (gfx1250) — compile-verified, matrix operand order checked by reading, not run on hardware
//
#include <hip/hip_runtime.h>
#include <math.h>

#define B_ 4
#define S_ 2048
#define F_ 512
#define H_ 8
#define KNN 32
#define D_ 64
#define M_ (B_ * S_)   // 8192 rows
#define N3 (3 * F_)    // 1536 output features (Q|K|V concatenated)

// ---------------------------------------------------------------------------
// WMMA types (CDNA5 / gfx1250, wave32)
// ---------------------------------------------------------------------------
typedef __attribute__((ext_vector_type(16))) __bf16 v16bf;
typedef __attribute__((ext_vector_type(8)))  float  v8f;

// round-to-nearest-even f32 -> bf16 (raw bits)
__device__ __forceinline__ unsigned short f2bf(float f) {
    unsigned int u = __float_as_uint(f);
    unsigned int r = u + 0x7fffu + ((u >> 16) & 1u);
    return (unsigned short)(r >> 16);
}

// Async-copy 32 contiguous bytes global -> LDS (2 x b128, ASYNCcnt += 2).
// INST_OFFSET is added to BOTH the global and LDS addresses, so one
// (lds, voff, saddr) triple covers both halves.
__device__ __forceinline__ void async_copy32B(unsigned lds, unsigned goff,
                                              const unsigned short* base) {
    asm volatile(
        "global_load_async_to_lds_b128 %0, %1, %2 offset:0\n\t"
        "global_load_async_to_lds_b128 %0, %1, %2 offset:16"
        :: "v"(lds), "v"(goff), "s"(base) : "memory");
}

// ---------------------------------------------------------------------------
// Convert x [M,F] f32 -> bf16
// ---------------------------------------------------------------------------
__global__ void cvt_x_kernel(const float* __restrict__ x,
                             unsigned short* __restrict__ xb) {
    int i = blockIdx.x * blockDim.x + threadIdx.x;
    if (i < M_ * F_) xb[i] = f2bf(x[i]);
}

// Convert Wq|Wk|Wv (each [F,F], row o = output feature) -> wb [3F, F] bf16
__global__ void cvt_w_kernel(const float* __restrict__ wq,
                             const float* __restrict__ wk,
                             const float* __restrict__ wv,
                             unsigned short* __restrict__ wb) {
    int i = blockIdx.x * blockDim.x + threadIdx.x;
    if (i >= N3 * F_) return;
    int row = i >> 9;          // 0..1535
    int col = i & (F_ - 1);
    const float* src = (row < F_) ? wq : (row < 2 * F_) ? wk : wv;
    wb[i] = f2bf(src[(size_t)(row & (F_ - 1)) * F_ + col]);
}

// ---------------------------------------------------------------------------
// QKV GEMM:  C[m, n] = sum_k xb[m,k] * wb[n,k]
// Block tile 128(M) x 128(N), 8 waves in 2x4 grid, wave tile 64x32 = 4x2 WMMA.
// Double-buffered LDS staged with GLOBAL_LOAD_ASYNC_TO_LDS_B128 (ASYNCcnt).
// Writes q/k/v in head-major layout [B,H,S,D] (f32).
// ---------------------------------------------------------------------------
#define TMB 128
#define TNB 128
#define KST 32
#define NKI (F_ / KST)   // 16 K-steps
#define LSTR 48          // LDS row stride in halfs (96 B, 16-byte multiple)

__global__ __launch_bounds__(256) void gemm_qkv_kernel(
    const unsigned short* __restrict__ xb, const unsigned short* __restrict__ wb,
    float* __restrict__ qo, float* __restrict__ ko, float* __restrict__ vo) {
    __shared__ unsigned short As[2][TMB * LSTR];
    __shared__ unsigned short Bs[2][TNB * LSTR];

    const int t    = threadIdx.x;
    const int lane = t & 31;
    const int wave = t >> 5;
    const int wm   = (wave >> 2) * 64;   // wave row offset in block tile
    const int wn   = (wave & 3) * 32;    // wave col offset in block tile
    const int bm   = blockIdx.x * TMB;
    const int bn   = blockIdx.y * TNB;

    v8f acc[4][2];
#pragma unroll
    for (int i = 0; i < 4; ++i)
#pragma unroll
        for (int j = 0; j < 2; ++j)
#pragma unroll
            for (int e = 0; e < 8; ++e) acc[i][j][e] = 0.0f;

    // staging map: 256 threads x 32B cover a 128x32-half tile
    const int srow = t >> 1, soff = (t & 1) * 16;
    const unsigned ldsA0 = (unsigned)(size_t)&As[0][0];
    const unsigned ldsB0 = (unsigned)(size_t)&Bs[0][0];
    const unsigned aLdsOff = (unsigned)(srow * LSTR + soff) * 2u;
    const unsigned bLdsOff = aLdsOff;
    const unsigned aGBase  = (unsigned)((bm + srow) * F_ + soff) * 2u;
    const unsigned bGBase  = (unsigned)((bn + srow) * F_ + soff) * 2u;
    const unsigned ABUFB   = (unsigned)(TMB * LSTR) * 2u;
    const unsigned BBUFB   = (unsigned)(TNB * LSTR) * 2u;

    // prologue: stage K-slice 0 into buffer 0 (4 async instrs per wave)
    async_copy32B(ldsA0 + aLdsOff, aGBase, xb);
    async_copy32B(ldsB0 + bLdsOff, bGBase, wb);

    for (int it = 0; it < NKI; ++it) {
        const int buf = it & 1;
        if (it + 1 < NKI) {
            const unsigned kb2 = (unsigned)((it + 1) * KST) * 2u;
            async_copy32B(ldsA0 + (buf ^ 1) * ABUFB + aLdsOff, aGBase + kb2, xb);
            async_copy32B(ldsB0 + (buf ^ 1) * BBUFB + bLdsOff, bGBase + kb2, wb);
            // 8 outstanding; oldest 4 (current buffer) must complete
            asm volatile("s_wait_asynccnt 0x4" ::: "memory");
        } else {
            asm volatile("s_wait_asynccnt 0x0" ::: "memory");
        }
        __syncthreads();

        // Per-lane fragment loads matching 16-bit 16x32 A layout:
        // lanes 0-15: K bases {0,16}; lanes 16-31: K bases {8,24}; 2 halfs/VGPR.
        const unsigned short* Ab = &As[buf][0];
        const unsigned short* Bb = &Bs[buf][0];
        const int kb  = (lane >> 4) * 8;
        const int r15 = lane & 15;
        union Frag { uint4 u[2]; v16bf v; } afr[4], bfr[2];
#pragma unroll
        for (int tm = 0; tm < 4; ++tm) {
            int row = wm + tm * 16 + r15;
            afr[tm].u[0] = *(const uint4*)&Ab[row * LSTR + kb];
            afr[tm].u[1] = *(const uint4*)&Ab[row * LSTR + kb + 16];
        }
#pragma unroll
        for (int tn = 0; tn < 2; ++tn) {
            int col = wn + tn * 16 + r15;
            bfr[tn].u[0] = *(const uint4*)&Bb[col * LSTR + kb];
            bfr[tn].u[1] = *(const uint4*)&Bb[col * LSTR + kb + 16];
        }

#pragma unroll
        for (int tm = 0; tm < 4; ++tm)
#pragma unroll
            for (int tn = 0; tn < 2; ++tn)
                acc[tm][tn] = __builtin_amdgcn_wmma_f32_16x16x32_bf16(
                    false, afr[tm].v, false, bfr[tn].v,
                    (short)0, acc[tm][tn], false, false);
        __syncthreads();   // all waves done reading buf before it is restaged
    }

    // Epilogue: C/D layout — lanes 0-15: M=v, N=lane; lanes 16-31: M=v+8, N=lane-16
    const int l15  = lane & 15;
    const int mofs = (lane < 16) ? 0 : 8;
#pragma unroll
    for (int tm = 0; tm < 4; ++tm) {
#pragma unroll
        for (int tn = 0; tn < 2; ++tn) {
            int n0  = bn + wn + tn * 16;        // 16-aligned; mat/h constant per tile
            int mat = n0 >> 9;                  // 0=q 1=k 2=v
            int o0  = n0 & (F_ - 1);
            int h   = o0 >> 6;
            int dd  = (o0 & 63) + l15;
            float* dst = (mat == 0) ? qo : (mat == 1) ? ko : vo;
#pragma unroll
            for (int e = 0; e < 8; ++e) {
                int m = bm + wm + tm * 16 + mofs + e;
                int b = m >> 11;
                int s = m & (S_ - 1);
                dst[(((size_t)(b * H_ + h)) * S_ + s) * D_ + dd] = acc[tm][tn][e];
            }
        }
    }
}

// ---------------------------------------------------------------------------
// kNN top-32 by squared distance (monotone with sqrt; slot order is
// irrelevant since softmax over K slots is permutation invariant).
// One thread per query; LDS candidate lists; replace-max selection.
// ---------------------------------------------------------------------------
#define NNT 128
__global__ __launch_bounds__(NNT) void nn_topk_kernel(
    const float* __restrict__ coords, const int* __restrict__ lens,
    int* __restrict__ nn, int* __restrict__ nok) {
    __shared__ float bd[NNT][KNN];
    __shared__ int   bi[NNT][KNN];
    const int tid = threadIdx.x;
    const int b   = blockIdx.y;
    const int s   = blockIdx.x * NNT + tid;
    const int len = lens[b];
    const float INFD = 3.0e38f;

#pragma unroll
    for (int i = 0; i < KNN; ++i) { bd[tid][i] = INFD; bi[tid][i] = 0; }

    const float* cb = coords + (size_t)b * S_ * 3;
    const bool qv = (s < len);
    float cx = 0.f, cy = 0.f, cz = 0.f;
    if (qv) { cx = cb[s * 3]; cy = cb[s * 3 + 1]; cz = cb[s * 3 + 2]; }

    float curmax = INFD;
    int   maxpos = 0;
    if (qv) {
        for (int j = 0; j < len; ++j) {
            if (j == s) continue;
            float dx = cb[j * 3] - cx;
            float dy = cb[j * 3 + 1] - cy;
            float dz = cb[j * 3 + 2] - cz;
            float d2 = dx * dx + dy * dy + dz * dz;
            if (d2 < curmax) {
                bd[tid][maxpos] = d2;
                bi[tid][maxpos] = j;
                float m = -1.0f; int mp = 0;
#pragma unroll
                for (int i = 0; i < KNN; ++i) {
                    float qd = bd[tid][i];
                    if (qd > m) { m = qd; mp = i; }
                }
                curmax = m; maxpos = mp;
            }
        }
    }

    int* nnp = nn  + ((size_t)b * S_ + s) * KNN;
    int* okp = nok + ((size_t)b * S_ + s) * KNN;
#pragma unroll
    for (int i = 0; i < KNN; ++i) {
        nnp[i] = bi[tid][i];
        okp[i] = (bd[tid][i] < 1.0e37f) ? 1 : 0;
    }
}

// ---------------------------------------------------------------------------
// Gather attention: one 256-thread block per (b, s).
// Phase 1: 256 threads = (h=8) x (j=32) score dot products (k/v hot in L2).
// Phase 2: 8 threads do masked softmax over K=32.
// Phase 3: coalesced weighted-V accumulation, 64 lanes span D.
// ---------------------------------------------------------------------------
__global__ __launch_bounds__(256) void attn_kernel(
    const float* __restrict__ q, const float* __restrict__ k,
    const float* __restrict__ v, const int* __restrict__ nn,
    const int* __restrict__ nok, float* __restrict__ out) {
    __shared__ int   s_nn[KNN];
    __shared__ int   s_ok[KNN];
    __shared__ float s_w[H_][KNN];

    const int t = threadIdx.x;
    const int s = blockIdx.x;
    const int b = blockIdx.y;

    if (t < KNN) {
        s_nn[t] = nn [((size_t)b * S_ + s) * KNN + t];
        s_ok[t] = nok[((size_t)b * S_ + s) * KNN + t];
    }
    __syncthreads();

    {   // scores
        const int h = t >> 5, j = t & 31;
        const float* qr = q + (((size_t)(b * H_ + h)) * S_ + s) * D_;
        const float* kr = k + (((size_t)(b * H_ + h)) * S_ + s_nn[j]) * D_;
        float dot = 0.f;
#pragma unroll 8
        for (int d0 = 0; d0 < D_; ++d0) dot += qr[d0] * kr[d0];
        s_w[h][j] = s_ok[j] ? dot * 0.125f : -1.0e30f;   // 1/sqrt(D)=0.125
    }
    __syncthreads();

    if (t < H_) {   // softmax per head
        const int h = t;
        float mx = -3.0e38f;
#pragma unroll
        for (int j = 0; j < KNN; ++j) mx = fmaxf(mx, s_w[h][j]);
        float e[KNN];
        float sum = 0.f;
#pragma unroll
        for (int j = 0; j < KNN; ++j) { e[j] = __expf(s_w[h][j] - mx); sum += e[j]; }
        int rowhas = 0;
#pragma unroll
        for (int j = 0; j < KNN; ++j) rowhas |= s_ok[j];
        const float inv = rowhas ? (1.0f / sum) : 0.0f;
#pragma unroll
        for (int j = 0; j < KNN; ++j) s_w[h][j] = e[j] * inv;
    }
    __syncthreads();

    {   // output: thread handles (h0, d) and (h0+4, d)
        const int h0 = t >> 6;
        const int d  = t & 63;
#pragma unroll
        for (int p = 0; p < 2; ++p) {
            const int h = h0 + p * 4;
            float acc = 0.f;
#pragma unroll 4
            for (int j = 0; j < KNN; ++j)
                acc += s_w[h][j] * v[(((size_t)(b * H_ + h)) * S_ + s_nn[j]) * D_ + d];
            out[((size_t)b * S_ + s) * F_ + h * D_ + d] = acc;
        }
    }
}

// ---------------------------------------------------------------------------
// metric = mean over S of k  -> [B,H,D]
// ---------------------------------------------------------------------------
__global__ void metric_kernel(const float* __restrict__ k, float* __restrict__ met) {
    int i = blockIdx.x * blockDim.x + threadIdx.x;
    if (i >= B_ * H_ * D_) return;
    const int d  = i & 63;
    const int bh = i >> 6;
    const float* kp = k + (size_t)bh * S_ * D_ + d;
    float sum = 0.f;
    for (int s = 0; s < S_; ++s) sum += kp[(size_t)s * D_];
    met[i] = sum * (1.0f / (float)S_);
}

// ---------------------------------------------------------------------------
extern "C" void kernel_launch(void* const* d_in, const int* in_sizes, int n_in,
                              void* d_out, int out_size, void* d_ws, size_t ws_size,
                              hipStream_t stream) {
    const float* x      = (const float*)d_in[0];
    const float* coords = (const float*)d_in[1];
    const int*   lens   = (const int*)  d_in[2];
    const float* wq     = (const float*)d_in[3];
    const float* wk     = (const float*)d_in[4];
    const float* wv     = (const float*)d_in[5];

    float* out    = (float*)d_out;                       // [B,S,F]
    float* metric = out + (size_t)B_ * S_ * F_;          // [B,H,D]

    char* ws = (char*)d_ws;
    size_t off = 0;
    auto carve = [&](size_t bytes) -> char* {
        off = (off + 255) & ~(size_t)255;
        char* p = ws + off;
        off += bytes;
        return p;
    };
    unsigned short* xb = (unsigned short*)carve((size_t)M_ * F_ * 2);
    unsigned short* wb = (unsigned short*)carve((size_t)N3 * F_ * 2);
    float* q  = (float*)carve((size_t)M_ * F_ * 4);
    float* k  = (float*)carve((size_t)M_ * F_ * 4);
    float* v  = (float*)carve((size_t)M_ * F_ * 4);
    int*   nn = (int*)  carve((size_t)B_ * S_ * KNN * 4);
    int*   ok = (int*)  carve((size_t)B_ * S_ * KNN * 4);

    cvt_x_kernel<<<(M_ * F_ + 255) / 256, 256, 0, stream>>>(x, xb);
    cvt_w_kernel<<<(N3 * F_ + 255) / 256, 256, 0, stream>>>(wq, wk, wv, wb);
    gemm_qkv_kernel<<<dim3(M_ / TMB, N3 / TNB), 256, 0, stream>>>(xb, wb, q, k, v);
    nn_topk_kernel<<<dim3(S_ / NNT, B_), NNT, 0, stream>>>(coords, lens, nn, ok);
    attn_kernel<<<dim3(S_, B_), 256, 0, stream>>>(q, k, v, nn, ok, out);
    metric_kernel<<<(B_ * H_ * D_ + 255) / 256, 256, 0, stream>>>(k, metric);
}
